// DuplicationRemovalNetwork_21706764714379
// MI455X (gfx1250) — matrix operand order repaired and checked by reading, not compiled, not run on hardware
//
#include <hip/hip_runtime.h>
#include <hip/hip_bf16.h>

// ---------------------------------------------------------------------------
// DuplicationRemovalNetwork for MI455X (gfx1250, wave32, WMMA)
//   B=32, K=1024, C=18, D=128, DQK=16
// All GEMMs / attention matmuls use v_wmma_f32_16x16x32_bf16 with operands
// pre-packed as bf16-pair dwords in the exact WMMA VGPR layout.
// ---------------------------------------------------------------------------

typedef __attribute__((ext_vector_type(16))) __bf16    v16bf;
typedef __attribute__((ext_vector_type(8)))  float     v8f;
typedef __attribute__((ext_vector_type(8)))  unsigned  v8u;

static constexpr int Bn   = 32;
static constexpr int Kn   = 1024;
static constexpr int Cn   = 18;
static constexpr int Dn   = 128;
static constexpr int DQK  = 16;

// Round-to-nearest-even pack of two f32 into one dword of two bf16.
static __device__ __forceinline__ unsigned bfpair(float lo, float hi) {
    unsigned ul = __float_as_uint(lo);
    unsigned uh = __float_as_uint(hi);
    unsigned rl = (ul + 0x7FFFu + ((ul >> 16) & 1u)) >> 16;
    unsigned rh = (uh + 0x7FFFu + ((uh >> 16) & 1u)) & 0xFFFF0000u;
    return rl | rh;
}

static __device__ __forceinline__ float fmaxsel(float a, float b) {
    return a > b ? a : b;              // no-NaN fast max (plain cmp+select)
}

static __device__ __forceinline__ v8f wmma_bf16(v8u au, v8u bu, v8f c) {
    v16bf a = __builtin_bit_cast(v16bf, au);
    v16bf b = __builtin_bit_cast(v16bf, bu);
    return __builtin_amdgcn_wmma_f32_16x16x32_bf16(
        false, a, false, b, (short)0, c, false, false);
}

// ---------------------------------------------------------------------------
// Kernel: class softmax max/argmax + bbox selection (pre-sort buffers)
// ---------------------------------------------------------------------------
__global__ void classify_kernel(const float* __restrict__ scores,
                                const float* __restrict__ center,
                                const float* __restrict__ sr,
                                const float* __restrict__ mean_size,
                                float* __restrict__ prob,
                                float* __restrict__ labelf,
                                float* __restrict__ bbox) {
    int idx = blockIdx.x * blockDim.x + threadIdx.x;   // b*K + k
    if (idx >= Bn * Kn) return;
    const float* s = scores + idx * Cn;
    float mx = s[0]; int am = 0;
    #pragma unroll
    for (int c = 1; c < Cn; ++c) { float v = s[c]; if (v > mx) { mx = v; am = c; } }
    float sum = 0.f;
    #pragma unroll
    for (int c = 0; c < Cn; ++c) sum += __expf(s[c] - mx);
    prob[idx]   = 1.f / sum;          // softmax max = exp(0)/sum
    labelf[idx] = (float)am;
    const float* ctr = center + idx * 3;
    const float* srr = sr + (idx * Cn + am) * 3;
    const float* msz = mean_size + am * 3;
    float* bb = bbox + idx * 6;
    bb[0] = ctr[0]; bb[1] = ctr[1]; bb[2] = ctr[2];
    bb[3] = srr[0] * msz[0] + msz[0];
    bb[4] = srr[1] * msz[1] + msz[1];
    bb[5] = srr[2] * msz[2] + msz[2];
}

// ---------------------------------------------------------------------------
// Kernel: per-batch stable descending sort (rank counting) + scatter/gather.
// Emits sorted features directly as packed bf16 pairs (GEMM-A layout).
// ---------------------------------------------------------------------------
__global__ __launch_bounds__(1024)
void sort_kernel(const float* __restrict__ prob,
                 const float* __restrict__ labelf,
                 const float* __restrict__ bbox,
                 const float* __restrict__ pfeat,      // [B, D, K]
                 float* __restrict__ sortedProb,       // [B, K]
                 float* __restrict__ outLabels,        // [B, K] (d_out)
                 float* __restrict__ outBbox,          // [B, K, 6] (d_out)
                 unsigned* __restrict__ sortedFpk) {   // [B*K, 64] bf16 pairs
    __shared__ float p[Kn];
    int b = blockIdx.x;
    int i = threadIdx.x;
    p[i] = prob[b * Kn + i];
    __syncthreads();
    float pi = p[i];
    int r = 0;
    for (int j = 0; j < Kn; ++j) {
        float pj = p[j];
        r += (pj > pi) || (pj == pi && j < i);   // stable descending rank
    }
    int src = b * Kn + i;
    int dst = b * Kn + r;
    sortedProb[dst] = pi;
    outLabels[dst]  = labelf[src];
    #pragma unroll
    for (int t = 0; t < 6; ++t) outBbox[dst * 6 + t] = bbox[src * 6 + t];
    // sorted_features[b][r][d] = proposal_features[b][d][i], packed in pairs
    const float* pf = pfeat + b * Dn * Kn + i;
    unsigned* dfp = sortedFpk + dst * (Dn / 2);
    #pragma unroll 4
    for (int d2 = 0; d2 < Dn / 2; ++d2)
        dfp[d2] = bfpair(pf[(2 * d2) * Kn], pf[(2 * d2 + 1) * Kn]);
}

// ---------------------------------------------------------------------------
// Kernel: sinusoidal rank embedding, emitted packed (bf16 pairs along D)
// ---------------------------------------------------------------------------
static __device__ __forceinline__ float emb_val(int k, int d) {
    int j = d & 63;                                    // half = 64
    float freq = __expf(-((float)j / 64.f) * 6.9077552789821f); // ln(1000)
    float v = (float)k * freq;
    return (d < 64) ? __sinf(v) : __cosf(v);
}
__global__ void rank_embed_kernel(unsigned* __restrict__ Epk) {
    int idx = blockIdx.x * blockDim.x + threadIdx.x;   // k*(D/2) + d2
    if (idx >= Kn * (Dn / 2)) return;
    int k = idx >> 6, d2 = idx & 63;
    Epk[idx] = bfpair(emb_val(k, 2 * d2), emb_val(k, 2 * d2 + 1));
}

// ---------------------------------------------------------------------------
// Kernel: generic f32 -> bf16-pair packer (pairs contiguous in memory)
// ---------------------------------------------------------------------------
__global__ void pack_pairs_kernel(const float* __restrict__ src,
                                  unsigned* __restrict__ dst, int n) {
    int idx = blockIdx.x * blockDim.x + threadIdx.x;
    if (idx >= n) return;
    dst[idx] = bfpair(src[2 * idx], src[2 * idx + 1]);
}

// ---------------------------------------------------------------------------
// Kernel: swizzle-pack V into the attention B-fragment layout:
//   Vpk[b][kt][half][r][d] = bfpair(V[b*K + kt*32 + half*16 + 2r][d],
//                                   V[b*K + kt*32 + half*16 + 2r+1][d])
// ---------------------------------------------------------------------------
__global__ void pack_v_kernel(const float* __restrict__ V,   // [B*K, D]
                              unsigned* __restrict__ Vpk) {  // [B*K*D/2]
    int idx = blockIdx.x * blockDim.x + threadIdx.x;
    if (idx >= Bn * Kn * Dn / 2) return;
    int d  = idx & 127;
    int r  = (idx >> 7) & 7;
    int h  = (idx >> 10) & 1;
    int kt = (idx >> 11) & 31;
    int b  = idx >> 16;
    int row0 = b * Kn + kt * 32 + h * 16 + 2 * r;
    Vpk[idx] = bfpair(V[row0 * Dn + d], V[(row0 + 1) * Dn + d]);
}

// ---------------------------------------------------------------------------
// Kernel: WMMA GEMM  C[M,N] = Apk[M,Kd] @ Wpk[N,Kd]^T + bias (+row_add)
// Operands are pre-packed bf16 pairs; one wave per 16x16 output tile.
// row_mask must be (pow2-1); row index into row_add = row & row_mask.
// ---------------------------------------------------------------------------
template <bool HAS_ADD>
__global__ __launch_bounds__(32)
void gemm_bias_wmma(const unsigned* __restrict__ Apk,   // [M, Kd/2]
                    const unsigned* __restrict__ Wpk,   // [N, Kd/2]
                    const float* __restrict__ bias,     // [N]
                    float* __restrict__ C, int N, int Kd,
                    const float* __restrict__ row_add, int row_mask) {
    const int lane = threadIdx.x;
    const int half = lane >> 4;
    const int l16  = lane & 15;
    const int tm   = blockIdx.x;
    const int tn   = blockIdx.y;
    const int kp2  = Kd >> 1;                 // pairs per row
    const unsigned* Ar = Apk + (tm * 16 + l16) * kp2;
    const unsigned* Wr = Wpk + (tn * 16 + l16) * kp2 + (half ? 8 : 0);
    const int a0 = half ? 8 : 0;
    const int a1 = half ? 12 : 4;

    v8f acc = {0.f, 0.f, 0.f, 0.f, 0.f, 0.f, 0.f, 0.f};
    for (int kp = 0; kp < kp2; kp += 16) {    // 32 K-values per iteration
        v8u au, bu;
        #pragma unroll
        for (int r = 0; r < 4; ++r) {
            au[r]     = Ar[kp + a0 + r];
            au[4 + r] = Ar[kp + a1 + r];
        }
        #pragma unroll
        for (int r = 0; r < 8; ++r) bu[r] = Wr[kp + r];
        acc = wmma_bf16(au, bu, acc);
    }

    const int col = tn * 16 + l16;            // C/D: N = lane%16
    const int rb  = tm * 16 + 8 * half;       // first row for this lane
    const float bv = bias[col];
    float addv[8];
    #pragma unroll
    for (int r = 0; r < 8; ++r)
        addv[r] = HAS_ADD ? row_add[((rb + r) & row_mask) * N + col] : 0.f;
    #pragma unroll
    for (int r = 0; r < 8; ++r)
        C[(rb + r) * N + col] = acc[r] + bv + addv[r];
}

// ---------------------------------------------------------------------------
// Kernel: fused self-attention + epilogue.
// One wave per (batch, 16-query tile). Full 16x1024 score strip in LDS (64KB),
// exact softmax (normalization folded into the P pack), P@V with WMMA,
// then gamma*out + x -> logit -> sigmoid*prob.
// ---------------------------------------------------------------------------
__global__ __launch_bounds__(32)
void attention_kernel(const unsigned* __restrict__ Qpk,  // [B*K, 8] bf16 pairs
                      const unsigned* __restrict__ Kpk,  // [B*K, 8] bf16 pairs
                      const unsigned* __restrict__ Vpk,  // swizzled, see pack_v
                      const float* __restrict__ X,       // [B*K, 128] f32
                      const float* __restrict__ sortedProb, // [B*K]
                      const float* __restrict__ gamma_p,
                      const float* __restrict__ logit_w, // [128]
                      const float* __restrict__ logit_b, // [1]
                      float* __restrict__ scores_out) {  // [B*K]
    __shared__ float S[16][Kn];        // 64 KB score strip / staging
    const int lane = threadIdx.x;
    const int half = lane >> 4;
    const int l16  = lane & 15;
    const int qt   = blockIdx.x;       // 0..63 query tile
    const int b    = blockIdx.y;       // 0..31 batch

    // Zero-pad mask for the K dimension beyond DQK=16: lanes 16-31 of the
    // B fragment carry K=16..31, which are pad. Loads are unconditional;
    // the pad is applied with an AND so the compiler cannot turn this into
    // EXEC-masked per-dword loads.
    const unsigned padmsk = half ? 0u : 0xFFFFFFFFu;

    // ---- Q fragment: 16 real K-values, padded to 32 with zeros ----
    v8u aq;
    {
        const unsigned* Qr = Qpk + (b * Kn + qt * 16 + l16) * 8;
        #pragma unroll
        for (int r = 0; r < 4; ++r) { aq[r] = Qr[r] & padmsk; aq[4 + r] = Qr[4 + r] & padmsk; }
    }

    // ---- S = Q @ K^T over all 64 key tiles ----
    for (int nt = 0; nt < 64; ++nt) {
        const unsigned* Kr = Kpk + (b * Kn + nt * 16 + l16) * 8;
        v8u bk;
        #pragma unroll
        for (int r = 0; r < 8; ++r)
            bk[r] = Kr[r] & padmsk;
        v8f c = {0.f, 0.f, 0.f, 0.f, 0.f, 0.f, 0.f, 0.f};
        c = wmma_bf16(aq, bk, c);
        #pragma unroll
        for (int r = 0; r < 8; ++r)
            S[r + 8 * half][nt * 16 + l16] = c[r];
    }
    __syncthreads();

    // ---- softmax (b128 LDS passes): lane handles row l16, column half ----
    // Leaves unnormalized exp() in LDS; 1/sum kept in a register and folded
    // into the P-fragment pack (the A-fragment lane for row m is l16==m).
    float inv;
    {
        float4* Sm = reinterpret_cast<float4*>(&S[l16][half * 512]); // 128 vec4
        float mx = -3.4e38f;
        for (int j = 0; j < 128; ++j) {
            float4 v = Sm[j];
            float m01 = fmaxsel(v.x, v.y);
            float m23 = fmaxsel(v.z, v.w);
            mx = fmaxsel(mx, fmaxsel(m01, m23));
        }
        mx = fmaxsel(mx, __shfl_xor(mx, 16, 32));
        float sum = 0.f;
        for (int j = 0; j < 128; ++j) {
            float4 v = Sm[j];
            v.x = __expf(v.x - mx); v.y = __expf(v.y - mx);
            v.z = __expf(v.z - mx); v.w = __expf(v.w - mx);
            Sm[j] = v;
            sum += (v.x + v.y) + (v.z + v.w);
        }
        sum += __shfl_xor(sum, 16, 32);
        inv = 1.f / sum;
    }
    __syncthreads();

    // ---- O = P @ V : 8 accumulators cover D=128 ----
    v8f o[8];
    #pragma unroll
    for (int t = 0; t < 8; ++t) o[t] = v8f{0.f,0.f,0.f,0.f,0.f,0.f,0.f,0.f};

    for (int kt = 0; kt < 32; ++kt) {          // 32 keys per iteration
        const int kb = kt * 32;
        // A fragment packed in-register from P (LDS, f32), normalized here
        v8u ap;
        {
            const int m   = l16;
            const int ka0 = kb + (half ? 16 : 0);
            const int ka1 = kb + (half ? 24 : 8);
            #pragma unroll
            for (int r = 0; r < 4; ++r) {
                ap[r]     = bfpair(S[m][ka0 + 2*r] * inv, S[m][ka0 + 2*r + 1] * inv);
                ap[4 + r] = bfpair(S[m][ka1 + 2*r] * inv, S[m][ka1 + 2*r + 1] * inv);
            }
        }
        // B fragments: straight dword loads from swizzle-packed V
        const unsigned* Vr = Vpk + (((b * 32 + kt) * 2 + half) * 8) * Dn + l16;
        #pragma unroll
        for (int t = 0; t < 8; ++t) {
            v8u bv;
            #pragma unroll
            for (int r = 0; r < 8; ++r)
                bv[r] = Vr[r * Dn + t * 16];
            o[t] = wmma_bf16(ap, bv, o[t]);
        }
    }

    // ---- stage O to LDS, fused epilogue ----
    __syncthreads();
    #pragma unroll
    for (int t = 0; t < 8; ++t)
        #pragma unroll
        for (int r = 0; r < 8; ++r)
            S[r + 8 * half][t * 16 + l16] = o[t][r];
    __syncthreads();

    if (lane < 16) {
        const int m = lane;
        const int grow = b * Kn + qt * 16 + m;
        const float g = gamma_p[0];
        const float4* Ov = reinterpret_cast<const float4*>(&S[m][0]);
        const float4* Xv = reinterpret_cast<const float4*>(X + grow * Dn);
        const float4* Lv = reinterpret_cast<const float4*>(logit_w);
        float acc = 0.f;
        #pragma unroll 4
        for (int d = 0; d < Dn / 4; ++d) {
            float4 ov = Ov[d], xv = Xv[d], lv = Lv[d];
            acc += (g * ov.x + xv.x) * lv.x + (g * ov.y + xv.y) * lv.y
                 + (g * ov.z + xv.z) * lv.z + (g * ov.w + xv.w) * lv.w;
        }
        acc += logit_b[0];
        float s1 = 1.f / (1.f + __expf(-acc));
        scores_out[grow] = s1 * sortedProb[grow];
    }
}

// ---------------------------------------------------------------------------
// Launch
// ---------------------------------------------------------------------------
extern "C" void kernel_launch(void* const* d_in, const int* in_sizes, int n_in,
                              void* d_out, int out_size, void* d_ws, size_t ws_size,
                              hipStream_t stream) {
    const float* pfeat     = (const float*)d_in[0];   // [B,D,K]
    const float* center    = (const float*)d_in[1];   // [B,K,3]
    const float* sr        = (const float*)d_in[2];   // [B,K,C,3]
    const float* cls       = (const float*)d_in[3];   // [B,K,C]
    const float* mean_size = (const float*)d_in[4];   // [C,3]
    const float* rank_w    = (const float*)d_in[5];   // [D,D]
    const float* rank_b    = (const float*)d_in[6];   // [D]
    const float* feat_w    = (const float*)d_in[7];   // [D,D]
    const float* feat_b    = (const float*)d_in[8];   // [D]
    const float* q_w       = (const float*)d_in[9];   // [16,D]
    const float* q_b       = (const float*)d_in[10];  // [16]
    const float* k_w       = (const float*)d_in[11];  // [16,D]
    const float* k_b       = (const float*)d_in[12];  // [16]
    const float* v_w       = (const float*)d_in[13];  // [D,D]
    const float* v_b       = (const float*)d_in[14];  // [D]
    const float* gamma     = (const float*)d_in[15];  // scalar
    const float* logit_w   = (const float*)d_in[16];  // [1,D]
    const float* logit_b   = (const float*)d_in[17];  // [1]

    float* out        = (float*)d_out;
    float* scores_out = out;                 // [B,K]
    float* labels_out = out + Bn * Kn;       // [B,K]
    float* bbox_out   = out + 2 * Bn * Kn;   // [B,K,6]

    // workspace layout (4-byte units)
    float* ws = (float*)d_ws;
    size_t off = 0;
    auto fbuf = [&](size_t n) { float* p = ws + off; off += n; return p; };
    auto ubuf = [&](size_t n) { unsigned* p = (unsigned*)(ws + off); off += n; return p; };

    float*    prob       = fbuf((size_t)Bn * Kn);
    float*    labelPre   = fbuf((size_t)Bn * Kn);
    float*    bboxPre    = fbuf((size_t)Bn * Kn * 6);
    float*    sortedProb = fbuf((size_t)Bn * Kn);
    unsigned* sortedFpk  = ubuf((size_t)Bn * Kn * Dn / 2);
    unsigned* Epk        = ubuf((size_t)Kn * Dn / 2);
    float*    R          = fbuf((size_t)Kn * Dn);
    float*    X          = fbuf((size_t)Bn * Kn * Dn);
    unsigned* Xpk        = ubuf((size_t)Bn * Kn * Dn / 2);
    float*    Qb         = fbuf((size_t)Bn * Kn * DQK);
    float*    Kb         = fbuf((size_t)Bn * Kn * DQK);
    float*    Vb         = fbuf((size_t)Bn * Kn * Dn);
    unsigned* Qpk        = ubuf((size_t)Bn * Kn * DQK / 2);
    unsigned* Kpk        = ubuf((size_t)Bn * Kn * DQK / 2);
    unsigned* Vpk        = ubuf((size_t)Bn * Kn * Dn / 2);
    unsigned* rankWpk    = ubuf((size_t)Dn * Dn / 2);
    unsigned* featWpk    = ubuf((size_t)Dn * Dn / 2);
    unsigned* qWpk       = ubuf((size_t)DQK * Dn / 2);
    unsigned* kWpk       = ubuf((size_t)DQK * Dn / 2);
    unsigned* vWpk       = ubuf((size_t)Dn * Dn / 2);
    (void)ws_size; (void)in_sizes; (void)n_in; (void)out_size;

    const int TPB = 256;
    auto blocks = [](int n, int t) { return (n + t - 1) / t; };

    // 0) pack weights once
    pack_pairs_kernel<<<blocks(Dn*Dn/2, TPB), TPB, 0, stream>>>(rank_w, rankWpk, Dn*Dn/2);
    pack_pairs_kernel<<<blocks(Dn*Dn/2, TPB), TPB, 0, stream>>>(feat_w, featWpk, Dn*Dn/2);
    pack_pairs_kernel<<<blocks(DQK*Dn/2, TPB), TPB, 0, stream>>>(q_w, qWpk, DQK*Dn/2);
    pack_pairs_kernel<<<blocks(DQK*Dn/2, TPB), TPB, 0, stream>>>(k_w, kWpk, DQK*Dn/2);
    pack_pairs_kernel<<<blocks(Dn*Dn/2, TPB), TPB, 0, stream>>>(v_w, vWpk, Dn*Dn/2);

    // 1) class softmax / argmax / bbox select
    classify_kernel<<<blocks(Bn*Kn, TPB), TPB, 0, stream>>>(
        cls, center, sr, mean_size, prob, labelPre, bboxPre);

    // 2) stable descending sort + scatter/gather (labels/bbox straight to d_out)
    sort_kernel<<<Bn, Kn, 0, stream>>>(
        prob, labelPre, bboxPre, pfeat, sortedProb, labels_out, bbox_out, sortedFpk);

    // 3) sinusoidal rank embedding (packed)
    rank_embed_kernel<<<blocks(Kn*Dn/2, TPB), TPB, 0, stream>>>(Epk);

    // 4) R = E @ rank_w^T + rank_b              [1024,128]
    gemm_bias_wmma<false><<<dim3(Kn/16, Dn/16), 32, 0, stream>>>(
        Epk, rankWpk, rank_b, R, Dn, Dn, nullptr, 0);

    // 5) X = sortedF @ feat_w^T + feat_b + R[k] [B*K,128]
    gemm_bias_wmma<true><<<dim3(Bn*Kn/16, Dn/16), 32, 0, stream>>>(
        sortedFpk, featWpk, feat_b, X, Dn, Dn, R, Kn - 1);
    pack_pairs_kernel<<<blocks(Bn*Kn*Dn/2, TPB), TPB, 0, stream>>>(X, Xpk, Bn*Kn*Dn/2);

    // 6) Q/K/V projections (+ re-pack for attention)
    gemm_bias_wmma<false><<<dim3(Bn*Kn/16, DQK/16), 32, 0, stream>>>(
        Xpk, qWpk, q_b, Qb, DQK, Dn, nullptr, 0);
    gemm_bias_wmma<false><<<dim3(Bn*Kn/16, DQK/16), 32, 0, stream>>>(
        Xpk, kWpk, k_b, Kb, DQK, Dn, nullptr, 0);
    gemm_bias_wmma<false><<<dim3(Bn*Kn/16, Dn/16), 32, 0, stream>>>(
        Xpk, vWpk, v_b, Vb, Dn, Dn, nullptr, 0);
    pack_pairs_kernel<<<blocks(Bn*Kn*DQK/2, TPB), TPB, 0, stream>>>(Qb, Qpk, Bn*Kn*DQK/2);
    pack_pairs_kernel<<<blocks(Bn*Kn*DQK/2, TPB), TPB, 0, stream>>>(Kb, Kpk, Bn*Kn*DQK/2);
    pack_v_kernel<<<blocks(Bn*Kn*Dn/2, TPB), TPB, 0, stream>>>(Vb, Vpk);

    // 7) fused attention + logit/sigmoid epilogue -> nms_scores
    attention_kernel<<<dim3(Kn/16, Bn), 32, 0, stream>>>(
        Qpk, Kpk, Vpk, X, sortedProb, gamma, logit_w, logit_b, scores_out);
}
